// LinearInt4_61856118997380
// MI455X (gfx1250) — 8-bit matrix operand order repaired and checked by reading, compiled, not run on hardware
//
#include <hip/hip_runtime.h>
#include <cstdint>


typedef int   v8i  __attribute__((ext_vector_type(8)));
typedef int   v16i __attribute__((ext_vector_type(16)));
typedef float v8f  __attribute__((ext_vector_type(8)));

namespace {
constexpr int Mtok = 4096;
constexpr int Kdim = 4096;
constexpr int Nout = 4096;
constexpr int K4   = 3968;
constexpr int NG   = 32;
}

union FragA8  { v8i  v; int4 d4[2]; };
union FragB8  { v8i  v; int4 d4[2]; };
union AccI    { v8i  v; int  s[8];  };
union FragA16 { v16i v; int4 d4[4]; };
union FragB16 { v16i v; int4 d4[4]; };
union AccF    { v8f  v; float s[8]; };

__device__ __forceinline__ uint32_t fp8_of_int4(int v) {
    const uint64_t LO = 0xB8C0C4C8CACCCED0ull;
    const uint64_t HI = 0x4E4C4A4844403800ull;
    const int idx = (v + 8) & 15;
    const uint64_t h = (idx & 8) ? HI : LO;
    return (uint32_t)((h >> ((idx & 7) * 8)) & 0xFF);
}

__global__ __launch_bounds__(256)
void pack_a_kernel(const int* __restrict__ norms, const int* __restrict__ outlier,
                   uint8_t* __restrict__ A8s)
{
    const int tx = blockIdx.x * blockDim.x + threadIdx.x;
    const int m  = blockIdx.y;
    const int k0 = tx * 8;
    uint32_t by[8];
    if (k0 < K4) {
        const int* p = norms + (size_t)m * K4 + k0;
        #pragma unroll
        for (int i = 0; i < 8; ++i) by[i] = fp8_of_int4(p[i]);
    } else {
        const int* p = outlier + (size_t)m * 128 + (k0 - K4);
        #pragma unroll
        for (int i = 0; i < 8; ++i) by[i] = (uint32_t)(p[i] & 0xFF);
    }
    uint2 packed;
    packed.x = by[0] | (by[1] << 8) | (by[2] << 16) | (by[3] << 24);
    packed.y = by[4] | (by[5] << 8) | (by[6] << 16) | (by[7] << 24);
    const int c = k0 >> 6, r = k0 & 63;
    const int doff = c * 64 + ((r >> 4) & 1) * 32 + ((r >> 5) & 1) * 16 + (r & 8);
    *(uint2*)(A8s + (size_t)m * Kdim + doff) = packed;
}

__global__ __launch_bounds__(256)
void pack_w_kernel(const int* __restrict__ w4, const int* __restrict__ w8,
                   uint8_t* __restrict__ W8)
{
    const int tx = blockIdx.x * blockDim.x + threadIdx.x;
    const int o  = blockIdx.y;
    if (tx < 992) {
        const int j = tx;
        int b0 = w4[(size_t)o * 1984 + 2 * j];
        int b1 = w4[(size_t)o * 1984 + 2 * j + 1];
        uint32_t lo0 = fp8_of_int4((b0 & 0xF) - 8), hi0 = fp8_of_int4(((b0 >> 4) & 0xF) - 8);
        uint32_t lo1 = fp8_of_int4((b1 & 0xF) - 8), hi1 = fp8_of_int4(((b1 >> 4) & 0xF) - 8);
        uint32_t packed = lo0 | (hi0 << 8) | (lo1 << 16) | (hi1 << 24);
        *(uint32_t*)(W8 + (size_t)o * Kdim + 4 * j) = packed;
    } else {
        const int i = tx - 992;
        const int* p = w8 + (size_t)o * 128 + 4 * i;
        uint32_t packed = (uint32_t)(p[0] & 0xFF)        | ((uint32_t)(p[1] & 0xFF) << 8) |
                          ((uint32_t)(p[2] & 0xFF) << 16) | ((uint32_t)(p[3] & 0xFF) << 24);
        *(uint32_t*)(W8 + (size_t)o * Kdim + K4 + 4 * i) = packed;
    }
}

__global__ __launch_bounds__(256)
void pack_scales_kernel(const float* __restrict__ norm_scales,
                        const float* __restrict__ scale_int4,
                        const float* __restrict__ outlier_scales,
                        const float* __restrict__ scale_int8,
                        float* __restrict__ SA_T, float* __restrict__ SB)
{
    const int idx = blockIdx.x * blockDim.x + threadIdx.x;
    const int nSB = NG * Nout;
    if (idx < nSB) {
        int g = idx / Nout, o = idx % Nout;
        SB[idx] = (g < NG - 1) ? scale_int4[(size_t)g * Nout + o] : scale_int8[o];
    } else {
        int j = idx - nSB;
        int g = j / Mtok, m = j % Mtok;
        SA_T[j] = (g < NG - 1) ? norm_scales[(size_t)m * (NG - 1) + g] : outlier_scales[m];
    }
}

__global__ __launch_bounds__(256)
void gemm_kernel(const uint8_t* __restrict__ A8s, const uint8_t* __restrict__ W8,
                 const float* __restrict__ SA_T, const float* __restrict__ SB,
                 _Float16* __restrict__ out)
{
    const int lane = threadIdx.x & 31;
    const int wave = threadIdx.x >> 5;
    const int half = lane >> 4;
    const int l16  = lane & 15;
    const int wm   = wave >> 1;
    const int wn   = wave & 1;
    const int m_wave = blockIdx.y * 128 + wm * 32;
    const int n_wave = blockIdx.x * 128 + wn * 64;

    float accf[2][4][8];
    #pragma unroll
    for (int mt = 0; mt < 2; ++mt)
        #pragma unroll
        for (int nt = 0; nt < 4; ++nt)
            #pragma unroll
            for (int r = 0; r < 8; ++r) accf[mt][nt][r] = 0.0f;

    const v8f czero = {};

    for (int g = 0; g < NG - 1; ++g) {
        const int kk = g * 128;

        float sb[4];
        FragB16 b[4];
        #pragma unroll
        for (int nt = 0; nt < 4; ++nt) {
            sb[nt] = SB[(size_t)g * Nout + n_wave + nt * 16 + l16];
            const uint8_t* wp = W8 + (size_t)(n_wave + nt * 16 + l16) * Kdim + kk + half * 16;
            b[nt].d4[0] = *(const int4*)(wp);
            b[nt].d4[1] = *(const int4*)(wp + 32);
            b[nt].d4[2] = *(const int4*)(wp + 64);
            b[nt].d4[3] = *(const int4*)(wp + 96);
            __builtin_prefetch(wp + 128, 0, 3);
        }

        #pragma unroll
        for (int mt = 0; mt < 2; ++mt) {
            const int row = m_wave + mt * 16 + l16;
            const uint8_t* ap = A8s + (size_t)row * Kdim + kk + half * 32;
            FragA16 a;
            a.d4[0] = *(const int4*)(ap);
            a.d4[1] = *(const int4*)(ap + 16);
            a.d4[2] = *(const int4*)(ap + 64);
            a.d4[3] = *(const int4*)(ap + 80);
            __builtin_prefetch(ap + 128, 0, 3);

            const int mrow = m_wave + mt * 16 + half * 8;
            const float4 sal = *(const float4*)(SA_T + (size_t)g * Mtok + mrow);
            const float4 sah = *(const float4*)(SA_T + (size_t)g * Mtok + mrow + 4);
            const float sa[8] = { sal.x, sal.y, sal.z, sal.w, sah.x, sah.y, sah.z, sah.w };

            #pragma unroll
            for (int nt = 0; nt < 4; ++nt) {
                AccF d;
                d.v = __builtin_amdgcn_wmma_f32_16x16x128_fp8_fp8(
                          a.v, b[nt].v, (short)0, czero, false, false);
                #pragma unroll
                for (int r = 0; r < 8; ++r)
                    accf[mt][nt][r] = fmaf(sa[r] * sb[nt], d.s[r], accf[mt][nt][r]);
            }
        }
    }

    {
        const int g  = NG - 1;
        const int kk = g * 128;

        float sb[4];
        FragB8 b[4][2];
        #pragma unroll
        for (int nt = 0; nt < 4; ++nt) {
            sb[nt] = SB[(size_t)g * Nout + n_wave + nt * 16 + l16];
            const uint8_t* wp = W8 + (size_t)(n_wave + nt * 16 + l16) * Kdim + kk + half * 16;
            b[nt][0].d4[0] = *(const int4*)(wp);
            b[nt][0].d4[1] = *(const int4*)(wp + 32);
            b[nt][1].d4[0] = *(const int4*)(wp + 64);
            b[nt][1].d4[1] = *(const int4*)(wp + 96);
        }

        #pragma unroll
        for (int mt = 0; mt < 2; ++mt) {
            const int row = m_wave + mt * 16 + l16;
            const uint8_t* ap = A8s + (size_t)row * Kdim + kk + half * 32;
            FragA8 a0, a1;
            a0.d4[0] = *(const int4*)(ap);
            a0.d4[1] = *(const int4*)(ap + 16);
            a1.d4[0] = *(const int4*)(ap + 64);
            a1.d4[1] = *(const int4*)(ap + 80);

            const int mrow = m_wave + mt * 16 + half * 8;
            const float4 sal = *(const float4*)(SA_T + (size_t)g * Mtok + mrow);
            const float4 sah = *(const float4*)(SA_T + (size_t)g * Mtok + mrow + 4);
            const float sa[8] = { sal.x, sal.y, sal.z, sal.w, sah.x, sah.y, sah.z, sah.w };

            #pragma unroll
            for (int nt = 0; nt < 4; ++nt) {
                AccI c;
                #pragma unroll
                for (int r = 0; r < 8; ++r) c.s[r] = 0;
                c.v = __builtin_amdgcn_wmma_i32_16x16x64_iu8(true, a0.v, true, b[nt][0].v, c.v, false, false);
                c.v = __builtin_amdgcn_wmma_i32_16x16x64_iu8(true, a1.v, true, b[nt][1].v, c.v, false, false);
                #pragma unroll
                for (int r = 0; r < 8; ++r)
                    accf[mt][nt][r] = fmaf(sa[r] * sb[nt], (float)c.s[r], accf[mt][nt][r]);
            }
        }
    }

    #pragma unroll
    for (int mt = 0; mt < 2; ++mt) {
        const int mrow = m_wave + mt * 16 + half * 8;
        #pragma unroll
        for (int nt = 0; nt < 4; ++nt) {
            const int col = n_wave + nt * 16 + l16;
            #pragma unroll
            for (int r = 0; r < 8; ++r)
                out[(size_t)(mrow + r) * Nout + col] = (_Float16)accf[mt][nt][r];
        }
    }
}

extern "C" void kernel_launch(void* const* d_in, const int* in_sizes, int n_in,
                              void* d_out, int out_size, void* d_ws, size_t ws_size,
                              hipStream_t stream)
{
    const int*   norms          = (const int*)d_in[0];
    const int*   weight_int4    = (const int*)d_in[1];
    const float* norm_scales    = (const float*)d_in[2];
    const float* scale_int4     = (const float*)d_in[3];
    const int*   outlier        = (const int*)d_in[4];
    const int*   weight_int8    = (const int*)d_in[5];
    const float* outlier_scales = (const float*)d_in[6];
    const float* scale_int8     = (const float*)d_in[7];
    _Float16*    out            = (_Float16*)d_out;

    char*    ws   = (char*)d_ws;
    uint8_t* A8s  = (uint8_t*)ws;
    uint8_t* W8   = (uint8_t*)(ws + (size_t)Mtok * Kdim);
    float*   SA_T = (float*)(ws + (size_t)Mtok * Kdim + (size_t)Nout * Kdim);
    float*   SB   = SA_T + (size_t)NG * Mtok;

    dim3 blk(256);
    pack_a_kernel<<<dim3(2, Mtok), blk, 0, stream>>>(norms, outlier, A8s);
    pack_w_kernel<<<dim3(4, Nout), blk, 0, stream>>>(weight_int4, weight_int8, W8);
    pack_scales_kernel<<<dim3((NG * Nout + NG * Mtok) / 256), blk, 0, stream>>>(
        norm_scales, scale_int4, outlier_scales, scale_int8, SA_T, SB);
    gemm_kernel<<<dim3(Nout / 128, Mtok / 128), blk, 0, stream>>>(A8s, W8, SA_T, SB, out);
}
